// NeighborhoodSelfAttentionBlock_26963804684923
// MI455X (gfx1250) — matrix operand order repaired and checked by reading, not compiled, not run on hardware
//
#include <hip/hip_runtime.h>
#include <hip/hip_fp16.h>

#define D_MODEL 384
#define N_HEADS 6
#define D_HEAD  64
#define QKV_N   1152
#define COND_D  768
#define KS      7
#define H_DIM   128
#define W_DIM   128
#define N_BATCH 4
#define N_PIX   (N_BATCH * H_DIM * W_DIM)   // 65536
#define GEMM_K  384                          // both GEMMs have K = 384

typedef __attribute__((ext_vector_type(16))) _Float16 v16h;
typedef __attribute__((ext_vector_type(8)))  _Float16 v8h;
typedef __attribute__((ext_vector_type(8)))  float    v8f;

// ---------------------------------------------------------------------------
// 1) gate[b][j] = 1 + sum_c cond[b][c] * w_cond[c][j]
// ---------------------------------------------------------------------------
__global__ void gate_kernel(const float* __restrict__ cond,
                            const float* __restrict__ wc,
                            float* __restrict__ gate) {
    int i = blockIdx.x * 256 + threadIdx.x;
    if (i >= N_BATCH * D_MODEL) return;
    int b = i / D_MODEL, j = i % D_MODEL;
    float acc = 1.0f;
    for (int c = 0; c < COND_D; ++c)
        acc += cond[b * COND_D + c] * wc[c * D_MODEL + j];
    gate[i] = acc;
}

// ---------------------------------------------------------------------------
// 2) W (K x N, row-major f32) -> WT (N x K, row-major f16)
// ---------------------------------------------------------------------------
__global__ void transpose_f2h(const float* __restrict__ w,
                              _Float16* __restrict__ wT, int K, int N) {
    int i = blockIdx.x * 256 + threadIdx.x;
    if (i >= K * N) return;
    int n = i / K, k = i % K;
    wT[i] = (_Float16)w[(size_t)k * N + n];
}

// ---------------------------------------------------------------------------
// 3) per-row RMS norm * gate, f32 -> f16. One block (128 thr) per row of 384.
// ---------------------------------------------------------------------------
__global__ __launch_bounds__(128) void rmsnorm_kernel(
    const float* __restrict__ x, const float* __restrict__ gate,
    _Float16* __restrict__ xn) {
    __shared__ float red[128];
    int row = blockIdx.x, t = threadIdx.x;
    int b = row >> 14;                           // row / (128*128)
    const float* xr = x + (size_t)row * D_MODEL;
    float v0 = xr[t], v1 = xr[t + 128], v2 = xr[t + 256];
    red[t] = v0 * v0 + v1 * v1 + v2 * v2;
    __syncthreads();
    for (int s = 64; s > 0; s >>= 1) {
        if (t < s) red[t] += red[t + s];
        __syncthreads();
    }
    float f = rsqrtf(red[0] * (1.0f / 384.0f) + 1e-6f);
    const float* g = gate + b * D_MODEL;
    _Float16* o = xn + (size_t)row * D_MODEL;
    o[t]       = (_Float16)(v0 * f * g[t]);
    o[t + 128] = (_Float16)(v1 * f * g[t + 128]);
    o[t + 256] = (_Float16)(v2 * f * g[t + 256]);
}

// ---------------------------------------------------------------------------
// 4/7) WMMA GEMM: out[M,N] = A[M,K] @ B[K,N], B supplied transposed (N x K),
// K == 384.  Block (8 waves) computes a 128(M) x 64(N) tile.
//
// The 64x384 f16 B-supertile (48 KB) is staged ONCE per block into LDS with
// CDNA5 async-to-LDS copies (ASYNCcnt), then reused by all 8 waves: B global
// traffic drops 8x and the K-loop issues only 2 global b128 (A) per 4 WMMAs.
//
// A frag (16-bit 16x32): lane<16 -> K {0..15}; lane>=16 ->
// K {16..31}: two contiguous 8-f16 (16B) loads per lane.
// B frag (16-bit 32x16): lane = column N, 16 contiguous K values
// (lanes<16: K 0..15, lanes>=16: K 16..31): one 32B LDS read per lane.
// Epilogue: C/D layout -> row = m0 + i + 8*(lane>>4), col = n0 + (lane&15).
// ---------------------------------------------------------------------------
__global__ __launch_bounds__(256) void gemm_wmma(
    const _Float16* __restrict__ A, const _Float16* __restrict__ BT,
    int M, int N, int K,
    _Float16* __restrict__ outh, float* __restrict__ outf,
    const float* __restrict__ skip) {
    __shared__ _Float16 bsh[64 * GEMM_K];      // 48 KB

    const int tid  = threadIdx.x;
    const int wid  = tid >> 5;
    const int lane = tid & 31;
    const int r    = lane & 15;
    const int half = lane >> 4;
    const int ntn  = N >> 6;
    const int mb   = blockIdx.x / ntn;
    const int nt   = blockIdx.x % ntn;
    const int m0   = (mb << 7) + (wid << 4);
    const int n0   = nt << 6;

    // ---- async stage of the 64 x K f16 B-supertile into LDS ----------------
    {
        const char* src = (const char*)(BT + (size_t)n0 * K);
        unsigned int dst0 = (unsigned int)(uintptr_t)(&bsh[0]);
        const int total16 = (64 * GEMM_K * 2) / 16;          // 3072 xfers of 16B
        for (int t = tid; t < total16; t += 256) {
            unsigned long long g = (unsigned long long)(uintptr_t)(src + (size_t)t * 16);
            unsigned int l = dst0 + (unsigned int)t * 16u;
            asm volatile("global_load_async_to_lds_b128 %0, %1, off"
                         :: "v"(l), "v"(g) : "memory");
        }
        asm volatile("s_wait_asynccnt 0x0" ::: "memory");
        __syncthreads();
    }

    // ---- K loop: A from global, B from LDS ---------------------------------
    const _Float16* arow = A + (size_t)(m0 + r) * K + half * 16;
    const _Float16* lb0  = &bsh[(size_t)( 0 + r) * GEMM_K + half * 16];
    const _Float16* lb1  = &bsh[(size_t)(16 + r) * GEMM_K + half * 16];
    const _Float16* lb2  = &bsh[(size_t)(32 + r) * GEMM_K + half * 16];
    const _Float16* lb3  = &bsh[(size_t)(48 + r) * GEMM_K + half * 16];

    v8f acc0 = {}, acc1 = {}, acc2 = {}, acc3 = {};
    for (int kc = 0; kc < GEMM_K; kc += 32) {
        v8h lo = *(const v8h*)(arow + kc);
        v8h hi = *(const v8h*)(arow + kc + 8);
        v16h a = __builtin_shufflevector(lo, hi, 0, 1, 2, 3, 4, 5, 6, 7,
                                                 8, 9, 10, 11, 12, 13, 14, 15);
        v16h vb0 = *(const v16h*)(lb0 + kc);
        v16h vb1 = *(const v16h*)(lb1 + kc);
        v16h vb2 = *(const v16h*)(lb2 + kc);
        v16h vb3 = *(const v16h*)(lb3 + kc);
        acc0 = __builtin_amdgcn_wmma_f32_16x16x32_f16(false, a, false, vb0,
                                                      (short)0, acc0, false, false);
        acc1 = __builtin_amdgcn_wmma_f32_16x16x32_f16(false, a, false, vb1,
                                                      (short)0, acc1, false, false);
        acc2 = __builtin_amdgcn_wmma_f32_16x16x32_f16(false, a, false, vb2,
                                                      (short)0, acc2, false, false);
        acc3 = __builtin_amdgcn_wmma_f32_16x16x32_f16(false, a, false, vb3,
                                                      (short)0, acc3, false, false);
    }

    // ---- epilogue ----------------------------------------------------------
    const int orow = m0 + (half << 3);
    const int oc   = n0 + r;
    if (outh) {
#pragma unroll
        for (int i = 0; i < 8; ++i) {
            size_t base = (size_t)(orow + i) * N + oc;
            outh[base]      = (_Float16)acc0[i];
            outh[base + 16] = (_Float16)acc1[i];
            outh[base + 32] = (_Float16)acc2[i];
            outh[base + 48] = (_Float16)acc3[i];
        }
    } else {
#pragma unroll
        for (int i = 0; i < 8; ++i) {
            size_t base = (size_t)(orow + i) * N + oc;
            outf[base]      = acc0[i] + skip[base];
            outf[base + 16] = acc1[i] + skip[base + 16];
            outf[base + 32] = acc2[i] + skip[base + 32];
            outf[base + 48] = acc3[i] + skip[base + 48];
        }
    }
}

// ---------------------------------------------------------------------------
// 5) q/k: L2 normalize * sqrt(scale), then rotary on dims (i, i+16), i<16.
// One thread per (pixel, head); handles both q and k in place.
// ---------------------------------------------------------------------------
__global__ __launch_bounds__(256) void qknorm_rope_kernel(
    _Float16* __restrict__ qkv, const float* __restrict__ pos,
    const float* __restrict__ freqs, const float* __restrict__ scale) {
    int t = blockIdx.x * 256 + threadIdx.x;
    if (t >= N_PIX * N_HEADS) return;
    int head = t % N_HEADS, pix = t / N_HEADS;
    float p0 = pos[(size_t)pix * 2], p1 = pos[(size_t)pix * 2 + 1];
    float cs[16], sn[16];
#pragma unroll
    for (int f = 0; f < 8; ++f) {
        float fr = freqs[head * 8 + f];
        float t0 = p0 * fr, t1 = p1 * fr;
        cs[f] = cosf(t0);     sn[f] = sinf(t0);
        cs[8 + f] = cosf(t1); sn[8 + f] = sinf(t1);
    }
    float sq = sqrtf(scale[head]);
#pragma unroll
    for (int part = 0; part < 2; ++part) {
        _Float16* base = qkv + (size_t)pix * QKV_N + part * D_MODEL + head * D_HEAD;
        float x[64];
        float ss = 0.f;
#pragma unroll
        for (int i = 0; i < 64; ++i) { x[i] = (float)base[i]; ss += x[i] * x[i]; }
        float sc = sq * rsqrtf(ss + 1e-6f);
#pragma unroll
        for (int i = 0; i < 16; ++i) {
            float x1 = x[i] * sc, x2 = x[i + 16] * sc;
            base[i]      = (_Float16)(x1 * cs[i] - x2 * sn[i]);
            base[i + 16] = (_Float16)(x2 * cs[i] + x1 * sn[i]);
        }
#pragma unroll
        for (int i = 32; i < 64; ++i) base[i] = (_Float16)(x[i] * sc);
    }
}

// ---------------------------------------------------------------------------
// 6) 7x7 neighborhood attention. One wave per (pixel, head); lane owns 2 dims.
// ---------------------------------------------------------------------------
__global__ __launch_bounds__(256) void attn_kernel(
    const _Float16* __restrict__ qkv, _Float16* __restrict__ out) {
    int gw   = blockIdx.x * 8 + (threadIdx.x >> 5);
    int lane = threadIdx.x & 31;
    int head = gw % N_HEADS;
    int pix  = gw / N_HEADS;
    int b = pix >> 14, rem = pix & 16383, h = rem >> 7, w = rem & 127;
    int sh = h - KS / 2; sh = sh < 0 ? 0 : (sh > H_DIM - KS ? H_DIM - KS : sh);
    int sw = w - KS / 2; sw = sw < 0 ? 0 : (sw > W_DIM - KS ? W_DIM - KS : sw);

    const _Float16* qb = qkv + (size_t)pix * QKV_N + head * D_HEAD;
    float q0 = (float)qb[2 * lane], q1 = (float)qb[2 * lane + 1];

    float e[KS * KS];
    float mx = -1e30f;
#pragma unroll
    for (int j = 0; j < KS * KS; ++j) {
        int ih = sh + j / KS, iw = sw + j % KS;
        int kp = (((b << 7) + ih) << 7) + iw;
        const _Float16* kb = qkv + (size_t)kp * QKV_N + D_MODEL + head * D_HEAD;
        float p = q0 * (float)kb[2 * lane] + q1 * (float)kb[2 * lane + 1];
#pragma unroll
        for (int m = 16; m >= 1; m >>= 1) p += __shfl_xor(p, m, 32);
        e[j] = p;
        mx = fmaxf(mx, p);
    }
    float sum = 0.f;
#pragma unroll
    for (int j = 0; j < KS * KS; ++j) { e[j] = __expf(e[j] - mx); sum += e[j]; }
    float inv = 1.f / sum;
    float a0 = 0.f, a1 = 0.f;
#pragma unroll
    for (int j = 0; j < KS * KS; ++j) {
        int ih = sh + j / KS, iw = sw + j % KS;
        int kp = (((b << 7) + ih) << 7) + iw;
        const _Float16* vb = qkv + (size_t)kp * QKV_N + 2 * D_MODEL + head * D_HEAD;
        a0 += e[j] * (float)vb[2 * lane];
        a1 += e[j] * (float)vb[2 * lane + 1];
    }
    size_t ob = (size_t)pix * D_MODEL + head * D_HEAD + 2 * lane;
    out[ob]     = (_Float16)(a0 * inv);
    out[ob + 1] = (_Float16)(a1 * inv);
}

// ---------------------------------------------------------------------------
// Launcher.  Workspace layout (~194 MB):
//   [0, 6K)        gate f32
//   [8K, ..)       wqkvT f16 (1152x384)     884736 B
//   [+884736, ..)  woutT f16 (384x384)      294912 B
//   [2M, +48M)     xn f16  (65536x384)   -> reused as attention output
//   [2M+48M, ..)   qkv f16 (65536x1152)     150994944 B
// ---------------------------------------------------------------------------
extern "C" void kernel_launch(void* const* d_in, const int* in_sizes, int n_in,
                              void* d_out, int out_size, void* d_ws, size_t ws_size,
                              hipStream_t stream) {
    const float* x      = (const float*)d_in[0];
    const float* pos    = (const float*)d_in[1];
    const float* cond   = (const float*)d_in[2];
    const float* w_cond = (const float*)d_in[3];
    const float* w_qkv  = (const float*)d_in[4];
    const float* scale  = (const float*)d_in[5];
    const float* freqs  = (const float*)d_in[6];
    const float* w_out  = (const float*)d_in[7];
    float* out = (float*)d_out;

    char* ws = (char*)d_ws;
    float*    gate  = (float*)ws;
    _Float16* wqkvT = (_Float16*)(ws + 8192);
    _Float16* woutT = (_Float16*)(ws + 8192 + 884736);
    _Float16* xn    = (_Float16*)(ws + (size_t)(2u << 20));
    _Float16* qkvh  = (_Float16*)(ws + (size_t)(2u << 20) +
                                  (size_t)N_PIX * D_MODEL * sizeof(_Float16));

    gate_kernel<<<(N_BATCH * D_MODEL + 255) / 256, 256, 0, stream>>>(cond, w_cond, gate);
    transpose_f2h<<<(D_MODEL * QKV_N + 255) / 256, 256, 0, stream>>>(w_qkv, wqkvT,
                                                                     D_MODEL, QKV_N);
    transpose_f2h<<<(D_MODEL * D_MODEL + 255) / 256, 256, 0, stream>>>(w_out, woutT,
                                                                       D_MODEL, D_MODEL);
    rmsnorm_kernel<<<N_PIX, 128, 0, stream>>>(x, gate, xn);

    // qkv = xn @ w_qkv : 65536 x 384 x 1152 ; block = 128x64 tile
    gemm_wmma<<<(N_PIX / 128) * (QKV_N / 64), 256, 0, stream>>>(
        xn, wqkvT, N_PIX, QKV_N, GEMM_K, qkvh, nullptr, nullptr);

    qknorm_rope_kernel<<<(N_PIX * N_HEADS + 255) / 256, 256, 0, stream>>>(
        qkvh, pos, freqs, scale);

    // one wave per (pixel, head): 65536*6 waves, 8 per block
    attn_kernel<<<N_PIX * N_HEADS / 8, 256, 0, stream>>>(qkvh, xn);

    // out = attn @ w_out + skip : 65536 x 384 x 384, f32 epilogue into d_out
    gemm_wmma<<<(N_PIX / 128) * (D_MODEL / 64), 256, 0, stream>>>(
        xn, woutT, N_PIX, D_MODEL, GEMM_K, nullptr, out, x);
}
